// BottleneckedEnsembleAttention_77137612636318
// MI455X (gfx1250) — matrix operand order repaired and checked by reading, not compiled, not run on hardware
//
#include <hip/hip_runtime.h>

typedef __attribute__((ext_vector_type(16))) _Float16 v16h;
typedef __attribute__((ext_vector_type(8)))  _Float16 v8h;
typedef __attribute__((ext_vector_type(8)))  float    v8f;
typedef __attribute__((ext_vector_type(4)))  float    v4f;
typedef __attribute__((ext_vector_type(4)))  int      v4i;

#define B_ 2
#define H_ 16
#define T_ 2048
#define D_ 1024
#define U_ 64

static __device__ __forceinline__ v16h cat16(v8h lo, v8h hi) {
  return __builtin_shufflevector(lo, hi, 0,1,2,3,4,5,6,7,8,9,10,11,12,13,14,15);
}
static __device__ __forceinline__ v8f wmma16(v16h a, v16h b, v8f c) {
  return __builtin_amdgcn_wmma_f32_16x16x32_f16(false, a, false, b, (short)0, c, false, false);
}
// SWAPX16 across the two 16-lane halves of a wave32 (xor=0x10, and=0x1f)
static __device__ __forceinline__ float swap16(float x) {
  return __int_as_float(__builtin_amdgcn_ds_swizzle(__float_as_int(x), 0x401F));
}
// YaRN (NTK-by-parts) inverse frequency; HEAD_DIM=64, theta=1e4, scale=4,
// train_len=2048, alpha=1, beta=32  =>  low=8, high=21, ramp denom=13.
static __device__ __forceinline__ float yarn_invfreq(int j) {
  float jf = (float)j;
  float pf = exp2f(jf * (13.287712379549449f / 32.0f)); // theta^(j/32)
  float inv_extra = 1.0f / pf;
  float inv_inter = 0.25f / pf;
  float ramp = fminf(fmaxf((jf - 8.0f) * (1.0f / 13.0f), 0.0f), 1.0f);
  return inv_inter * ramp + inv_extra * (1.0f - ramp);
}

// ---------------------------------------------------------------- kernel 0
// Convert + transpose projection weights to f16 in WMMA-B-friendly layouts:
//   WqT/WkT/WvT : [h][u=64][d=1024]   (column-major weights)
//   WoT         : [h][d=1024][u=64]
__global__ void __launch_bounds__(256) prep_weights_k(
    const float* __restrict__ qp, const float* __restrict__ kp,
    const float* __restrict__ vp, const float* __restrict__ opj,
    _Float16* __restrict__ WqT, _Float16* __restrict__ WkT,
    _Float16* __restrict__ WvT, _Float16* __restrict__ WoT)
{
  const int N1 = H_ * D_ * U_; // 1048576
  int i = blockIdx.x * blockDim.x + threadIdx.x;   // grid covers 4*N1 exactly
  int which = i / N1;
  int o = i - which * N1;
  if (which < 3) {
    int h = o >> 16;            // / (64*1024)
    int rem = o & 65535;
    int u = rem >> 10;
    int d = rem & 1023;
    const float* src = (which == 0) ? qp : (which == 1) ? kp : vp;
    _Float16* dst = (which == 0) ? WqT : (which == 1) ? WkT : WvT;
    dst[o] = (_Float16)src[(size_t)((h << 10) + d) * U_ + u];
  } else {
    int h = o >> 16;
    int rem = o & 65535;
    int d = rem >> 6;
    int u = rem & 63;
    WoT[o] = (_Float16)opj[(size_t)((h << 6) + u) * D_ + d];
  }
}

// ---------------------------------------------------------------- kernel 1
// One wave per (b,h, 16-token tile): Q,K,V = X @ W (f16 WMMA, fp32 acc),
// YaRN RoPE on Q,K (attention scale folded into Q). Q,K row-major f16;
// V stored transposed Vt[u][T] f16 for direct use as WMMA A-operand later.
__global__ void __launch_bounds__(256) qkv_rope_k(
    const float* __restrict__ X, const int* __restrict__ pos,
    const _Float16* __restrict__ WqT, const _Float16* __restrict__ WkT,
    const _Float16* __restrict__ WvT,
    _Float16* __restrict__ Q, _Float16* __restrict__ K, _Float16* __restrict__ Vt)
{
  int wave = blockIdx.x * (blockDim.x >> 5) + (threadIdx.x >> 5);
  int lane = threadIdx.x & 31;
  int tile = wave & 127;
  int bh   = wave >> 7;
  if (bh >= B_ * H_) return;
  int h = bh & (H_ - 1);
  int n = lane & 15, hi = lane >> 4;

  const float* xrow = X + ((size_t)bh * T_ + tile * 16 + n) * D_;
  const _Float16* wq = WqT + (size_t)h * U_ * D_;
  const _Float16* wk = WkT + (size_t)h * U_ * D_;
  const _Float16* wv = WvT + (size_t)h * U_ * D_;

  v8f aq[4] = {}, ak[4] = {}, av[4] = {};

  for (int kk = 0; kk < D_; kk += 32) {
    // A fragment (16x32 f16): elems 0..15 = K kk+hi*16..
    v4f f0 = *(const v4f*)(xrow + kk + hi * 16);
    v4f f1 = *(const v4f*)(xrow + kk + hi * 16 + 4);
    v4f f2 = *(const v4f*)(xrow + kk + hi * 16 + 8);
    v4f f3 = *(const v4f*)(xrow + kk + hi * 16 + 12);
    v16h a;
    #pragma unroll
    for (int i = 0; i < 4; ++i) {
      a[i]      = (_Float16)f0[i];
      a[4 + i]  = (_Float16)f1[i];
      a[8 + i]  = (_Float16)f2[i];
      a[12 + i] = (_Float16)f3[i];
    }
    int koff = kk + hi * 16;
    #pragma unroll
    for (int nt = 0; nt < 4; ++nt) {
      size_t wo = (size_t)(nt * 16 + n) * D_ + koff;
      aq[nt] = wmma16(a, *(const v16h*)(wq + wo), aq[nt]);
      ak[nt] = wmma16(a, *(const v16h*)(wk + wo), ak[nt]);
      av[nt] = wmma16(a, *(const v16h*)(wv + wo), av[nt]);
    }
  }

  // RoPE: C layout => lane holds (tok = tile*16 + r + hi*8, u = nt*16 + n)
  const int* prow = pos + (size_t)bh * T_ + tile * 16 + hi * 8;
  v4i pA = *(const v4i*)(prow);
  v4i pB = *(const v4i*)(prow + 4);
  float if0 = yarn_invfreq(n);        // u = n   and u = n+32
  float if1 = yarn_invfreq(n + 16);   // u = n+16 and u = n+48
  const float sc = 0.14232867951399863f;  // (0.1*ln4 + 1) / sqrt(64)

  _Float16* qb = Q + (size_t)bh * T_ * U_;
  _Float16* kb = K + (size_t)bh * T_ * U_;

  #pragma unroll
  for (int r = 0; r < 8; ++r) {
    float pp = (float)((r < 4) ? pA[r] : pB[r - 4]);
    float s0, c0, s1, c1;
    sincosf(pp * if0, &s0, &c0);
    sincosf(pp * if1, &s1, &c1);
    float q0 = aq[0][r], q1 = aq[1][r], q2 = aq[2][r], q3 = aq[3][r];
    float k0 = ak[0][r], k1 = ak[1][r], k2 = ak[2][r], k3 = ak[3][r];
    float rq0 = q0 * c0 - q2 * s0, rq2 = q2 * c0 + q0 * s0;
    float rq1 = q1 * c1 - q3 * s1, rq3 = q3 * c1 + q1 * s1;
    float rk0 = k0 * c0 - k2 * s0, rk2 = k2 * c0 + k0 * s0;
    float rk1 = k1 * c1 - k3 * s1, rk3 = k3 * c1 + k1 * s1;
    int tok = tile * 16 + hi * 8 + r;
    _Float16* qr = qb + (size_t)tok * U_ + n;
    qr[0]  = (_Float16)(rq0 * sc);
    qr[16] = (_Float16)(rq1 * sc);
    qr[32] = (_Float16)(rq2 * sc);
    qr[48] = (_Float16)(rq3 * sc);
    _Float16* kr = kb + (size_t)tok * U_ + n;
    kr[0]  = (_Float16)rk0;
    kr[16] = (_Float16)rk1;
    kr[32] = (_Float16)rk2;
    kr[48] = (_Float16)rk3;
  }
  // V transposed: Vt[u][tok]; r runs over consecutive tokens -> packed b128 store
  _Float16* vtb = Vt + (size_t)bh * U_ * T_;
  #pragma unroll
  for (int nt = 0; nt < 4; ++nt) {
    v8h pk;
    #pragma unroll
    for (int r = 0; r < 8; ++r) pk[r] = (_Float16)av[nt][r];
    *(v8h*)(vtb + (size_t)(nt * 16 + n) * T_ + tile * 16 + hi * 8) = pk;
  }
}

// ---------------------------------------------------------------- kernel 2
// Flash attention, one wave per (b,h, 16-query tile). Computes S^T = K@Q^T
// (softmax per-lane), P^T via LDS becomes the B operand of O^T = V^T @ P^T.
__global__ void __launch_bounds__(256) attn_k(
    const _Float16* __restrict__ Q, const _Float16* __restrict__ K,
    const _Float16* __restrict__ Vt, const unsigned char* __restrict__ amask,
    _Float16* __restrict__ Att)
{
  __shared__ __align__(32) _Float16 sP[8 * 512];  // per-wave 16q x 32k f16
  int wv = threadIdx.x >> 5, lane = threadIdx.x & 31;
  int wave = blockIdx.x * 8 + wv;
  int qt = wave & 127;
  int bh = wave >> 7;
  if (bh >= B_ * H_) return;
  int n = lane & 15, hi = lane >> 4;
  int qbase = qt * 16;

  const _Float16* q  = Q  + (size_t)bh * T_ * U_;
  const _Float16* k  = K  + (size_t)bh * T_ * U_;
  const _Float16* vt = Vt + (size_t)bh * U_ * T_;
  const unsigned char* am = amask + (size_t)bh * T_;
  _Float16* att = Att + (size_t)bh * T_ * U_;
  _Float16* lp = sP + wv * 512;

  // Q as B operand (lane = query column, dims hi*8..+7 and 16+hi*8..+7 of each 32)
  const _Float16* qrow = q + (size_t)(qbase + n) * U_;
  v16h bq0 = cat16(*(const v8h*)(qrow + hi * 8),      *(const v8h*)(qrow + 16 + hi * 8));
  v16h bq1 = cat16(*(const v8h*)(qrow + 32 + hi * 8), *(const v8h*)(qrow + 48 + hi * 8));
  int qglob = qbase + n;
  float qact = am[qglob] ? 1.0f : 0.0f;

  v8f o0 = {}, o1 = {}, o2 = {}, o3 = {};
  const float NEG = -3.0e38f;
  float m = NEG, ssum = 0.0f;

  for (int kblk = 0; kblk < qbase + 16; kblk += 32) {
    // ---- S^T tiles: A = K rows (keys), B = Q^T
    v8f st0 = {}, st1 = {};
    const _Float16* kr0 = k + (size_t)(kblk + n) * U_;
    const _Float16* kr1 = kr0 + 16 * U_;
    __builtin_prefetch(kr0 + 32 * U_, 0, 1);
    st0 = wmma16(cat16(*(const v8h*)(kr0 + hi * 8),      *(const v8h*)(kr0 + 16 + hi * 8)), bq0, st0);
    st0 = wmma16(cat16(*(const v8h*)(kr0 + 32 + hi * 8), *(const v8h*)(kr0 + 48 + hi * 8)), bq1, st0);
    st1 = wmma16(cat16(*(const v8h*)(kr1 + hi * 8),      *(const v8h*)(kr1 + 16 + hi * 8)), bq0, st1);
    st1 = wmma16(cat16(*(const v8h*)(kr1 + 32 + hi * 8), *(const v8h*)(kr1 + 48 + hi * 8)), bq1, st1);

    // ---- mask + online softmax (lane owns query n; keys = kblk + t*16 + hi*8 + r)
    unsigned long long km0 = *(const unsigned long long*)(am + kblk + hi * 8);
    unsigned long long km1 = *(const unsigned long long*)(am + kblk + 16 + hi * 8);
    float s0[8], s1[8];
    float mloc = NEG;
    #pragma unroll
    for (int r = 0; r < 8; ++r) {
      int key0 = kblk + hi * 8 + r;
      int key1 = kblk + 16 + hi * 8 + r;
      bool ok0 = (key0 <= qglob) && (((km0 >> (8 * r)) & 0xffULL) != 0);
      bool ok1 = (key1 <= qglob) && (((km1 >> (8 * r)) & 0xffULL) != 0);
      s0[r] = ok0 ? st0[r] : NEG;
      s1[r] = ok1 ? st1[r] : NEG;
      mloc = fmaxf(mloc, fmaxf(s0[r], s1[r]));
    }
    float mblk = fmaxf(mloc, swap16(mloc));
    float mnew = fmaxf(m, mblk);
    float fac = expf(m - mnew);
    m = mnew;
    ssum *= fac;
    o0 *= fac; o1 *= fac; o2 *= fac; o3 *= fac;

    float sl = 0.0f;
    v8h p0, p1;
    #pragma unroll
    for (int r = 0; r < 8; ++r) {
      float e0 = expf(s0[r] - mnew);
      float e1 = expf(s1[r] - mnew);
      sl += e0 + e1;
      p0[r] = (_Float16)e0;
      p1[r] = (_Float16)e1;
    }
    ssum += sl + swap16(sl);

    // ---- P^T through LDS: store [q][key], reload as B operand
    *(v8h*)(lp + n * 32 + hi * 8)      = p0;
    *(v8h*)(lp + n * 32 + 16 + hi * 8) = p1;
    asm volatile("s_wait_dscnt 0" ::: "memory");
    v16h bp = cat16(*(const v8h*)(lp + n * 32 + hi * 8), *(const v8h*)(lp + n * 32 + 16 + hi * 8));

    // ---- O^T += V^T @ P^T  (A = Vt rows, keys hi*8..+7 and 16+hi*8..+7)
    const _Float16* vrow = vt + (size_t)n * T_ + kblk;
    o0 = wmma16(cat16(*(const v8h*)(vrow + hi * 8), *(const v8h*)(vrow + 16 + hi * 8)), bp, o0);
    vrow += 16 * T_;
    o1 = wmma16(cat16(*(const v8h*)(vrow + hi * 8), *(const v8h*)(vrow + 16 + hi * 8)), bp, o1);
    vrow += 16 * T_;
    o2 = wmma16(cat16(*(const v8h*)(vrow + hi * 8), *(const v8h*)(vrow + 16 + hi * 8)), bp, o2);
    vrow += 16 * T_;
    o3 = wmma16(cat16(*(const v8h*)(vrow + hi * 8), *(const v8h*)(vrow + 16 + hi * 8)), bp, o3);
  }

  float inv = qact / ssum;  // zero inactive query rows like the reference
  o0 *= inv; o1 *= inv; o2 *= inv; o3 *= inv;

  // store attended row-major [tok][u]; lane -> (tok = qbase+n, u = t*16 + hi*8 + r)
  _Float16* arow = att + (size_t)(qbase + n) * U_ + hi * 8;
  v8f oo[4] = {o0, o1, o2, o3};
  #pragma unroll
  for (int t = 0; t < 4; ++t) {
    v8h pk;
    #pragma unroll
    for (int r = 0; r < 8; ++r) pk[r] = (_Float16)oo[t][r];
    *(v8h*)(arow + t * 16) = pk;
  }
}

// ---------------------------------------------------------------- kernel 3
// out = attended @ Wo : one wave per (b,h, 16-token tile, 128-dim chunk)
__global__ void __launch_bounds__(256) oproj_k(
    const _Float16* __restrict__ At, const _Float16* __restrict__ WoT,
    float* __restrict__ out)
{
  int wave = blockIdx.x * 8 + (threadIdx.x >> 5);
  int lane = threadIdx.x & 31;
  int nch = wave & 7;
  int qt  = (wave >> 3) & 127;
  int bh  = wave >> 10;
  if (bh >= B_ * H_) return;
  int h = bh & (H_ - 1);
  int n = lane & 15, hi = lane >> 4;
  int qbase = qt * 16;

  const _Float16* arow = At + ((size_t)bh * T_ + qbase + n) * U_;
  v16h a0 = cat16(*(const v8h*)(arow + hi * 8),      *(const v8h*)(arow + 16 + hi * 8));
  v16h a1 = cat16(*(const v8h*)(arow + 32 + hi * 8), *(const v8h*)(arow + 48 + hi * 8));
  const _Float16* wo = WoT + (size_t)h * D_ * U_;
  float* orow = out + ((size_t)bh * T_ + qbase) * D_;

  #pragma unroll
  for (int nt = 0; nt < 8; ++nt) {
    int ntg = nch * 8 + nt;
    const _Float16* wcol = wo + (size_t)(ntg * 16 + n) * U_;
    v8f acc = {};
    acc = wmma16(a0, cat16(*(const v8h*)(wcol + hi * 8), *(const v8h*)(wcol + 16 + hi * 8)), acc);
    acc = wmma16(a1, cat16(*(const v8h*)(wcol + 32 + hi * 8), *(const v8h*)(wcol + 48 + hi * 8)), acc);
    #pragma unroll
    for (int r = 0; r < 8; ++r)
      orow[(size_t)(r + hi * 8) * D_ + ntg * 16 + n] = acc[r];
  }
}

// ----------------------------------------------------------------
extern "C" void kernel_launch(void* const* d_in, const int* in_sizes, int n_in,
                              void* d_out, int out_size, void* d_ws, size_t ws_size,
                              hipStream_t stream) {
  (void)in_sizes; (void)n_in; (void)out_size; (void)ws_size;
  const float* emb = (const float*)d_in[0];
  const int*   pos = (const int*)d_in[1];
  const unsigned char* am = (const unsigned char*)d_in[2];
  const float* qp = (const float*)d_in[3];
  const float* kp = (const float*)d_in[4];
  const float* vp = (const float*)d_in[5];
  const float* op = (const float*)d_in[6];
  float* out = (float*)d_out;

  char* ws = (char*)d_ws;
  const size_t MB = 1ull << 20;
  _Float16* WqT = (_Float16*)(ws + 0 * MB);   // 2 MB
  _Float16* WkT = (_Float16*)(ws + 2 * MB);   // 2 MB
  _Float16* WvT = (_Float16*)(ws + 4 * MB);   // 2 MB
  _Float16* WoT = (_Float16*)(ws + 6 * MB);   // 2 MB
  _Float16* Q   = (_Float16*)(ws + 8 * MB);   // 8 MB
  _Float16* K   = (_Float16*)(ws + 16 * MB);  // 8 MB
  _Float16* Vt  = (_Float16*)(ws + 24 * MB);  // 8 MB
  _Float16* At  = (_Float16*)(ws + 32 * MB);  // 8 MB

  prep_weights_k<<<16384, 256, 0, stream>>>(qp, kp, vp, op, WqT, WkT, WvT, WoT);
  qkv_rope_k<<<512, 256, 0, stream>>>(emb, pos, WqT, WkT, WvT, Q, K, Vt);
  attn_k<<<512, 256, 0, stream>>>(Q, K, Vt, am, At);
  oproj_k<<<4096, 256, 0, stream>>>(At, WoT, out);
}
